// MultiHeadAttention_83176336655001
// MI455X (gfx1250) — matrix operand order repaired and checked by reading, not compiled, not run on hardware
//
#include <hip/hip_runtime.h>
#include <hip/hip_bf16.h>

// ---------------------------------------------------------------------------
// MultiHeadAttention forward for MI455X (gfx1250, wave32, WMMA).
// B=2, S=2048, D=1024, H=16, HD=64.  All matmuls on v_wmma_f32_16x16x32_bf16
// (fp32 accumulate).  Attention = flash-style online softmax (no SxS tensor).
// GEMMs: 32x64 wave tiles (2 A-frags x 4 B-frags = 8 WMMA / K-step); no
// manual double-buffering (round-2 showed it spills accumulators to scratch —
// the compiler already overlaps loads with partial s_wait_loadcnt on its own).
// ---------------------------------------------------------------------------

#define BB 2
#define SS 2048
#define DD 1024
#define HH 16
#define HD 64
#define MM (BB * SS)          // 4096 rows
#define ATT_SCALE 0.125f      // 64^-0.5

typedef __attribute__((ext_vector_type(16))) __bf16 v16bf;
typedef __attribute__((ext_vector_type(8)))  __bf16 v8bf;
typedef __attribute__((ext_vector_type(8)))  float  v8f;

static __device__ __forceinline__ v8f wmma_bf16(v16bf a, v16bf b, v8f c) {
    return __builtin_amdgcn_wmma_f32_16x16x32_bf16(
        false, a, false, b, (short)0, c, false, false);
}

// A-matrix 16x32 bf16 fragment (row-major, ld = row stride in elems).
// lanes 0-15: rows m0..m0+15, K 0..15; lanes 16-31: K 16..31.
static __device__ __forceinline__ v16bf load_a_frag(const __bf16* base, int ld,
                                                    int m0, int k0, int lane) {
    int row = m0 + (lane & 15);
    int kb  = k0 + ((lane >> 4) << 4);
    const v8bf* p0 = reinterpret_cast<const v8bf*>(base + (size_t)row * ld + kb);
    const v8bf* p1 = reinterpret_cast<const v8bf*>(base + (size_t)row * ld + kb + 8);
    v16bf a;
#pragma unroll
    for (int i = 0; i < 8; ++i) { a[i] = (*p0)[i]; a[i + 8] = (*p1)[i]; }
    return a;
}

// B-matrix 32x16 fragment where column n of B is row n of `base` (X * base^T).
// lanes 0-15: col n0+lane K k0..k0+15; lanes 16-31: same col K k0+16..k0+31.
static __device__ __forceinline__ v16bf load_b_frag(const __bf16* base, int ld,
                                                    int n0, int k0, int lane) {
    int row = n0 + (lane & 15);
    int kb  = k0 + ((lane >> 4) << 4);
    return *reinterpret_cast<const v16bf*>(base + (size_t)row * ld + kb);
}

// ---------------------------------------------------------------------------
// Staging: fp32 -> bf16, and rpr gather by positions
// ---------------------------------------------------------------------------
__global__ void cvt_f32_bf16(const float* __restrict__ in,
                             __bf16* __restrict__ out, int n) {
    int i = blockIdx.x * blockDim.x + threadIdx.x;
    if (i < n) out[i] = (__bf16)in[i];
}

__global__ void gather_rpr(const float* __restrict__ rpr,
                           const int* __restrict__ positions,
                           __bf16* __restrict__ out, int n /* B*S*HD */) {
    int i = blockIdx.x * blockDim.x + threadIdx.x;
    if (i >= n) return;
    int bs = i >> 6;
    int d  = i & 63;
    int pos = positions[bs];
    out[i] = (__bf16)rpr[(size_t)pos * HD + d];
}

// ---------------------------------------------------------------------------
// QKV projection GEMM: Y = X(4096x1024) * W^T(1024x1024), bf16 in/out.
// 32x64 wave tile; one B-frag streamed through 2 WMMAs at a time to keep
// VGPR pressure low (no spills).
// vmode==0: store (B,H,S,HD);  vmode==1: store (B,H,HD,S) (V transposed).
// ---------------------------------------------------------------------------
__global__ void gemm_qkv(const __bf16* __restrict__ X,
                         const __bf16* __restrict__ W,
                         __bf16* __restrict__ Y, int vmode) {
    int wave = threadIdx.x >> 5;
    int lane = threadIdx.x & 31;
    int gw   = blockIdx.x * (blockDim.x >> 5) + wave;
    int m0   = (gw >> 4) << 5;   // 128 M-tiles of 32
    int n0   = (gw & 15) << 6;   // 16  N-tiles of 64

    v8f acc0[4] = {}, acc1[4] = {};
    for (int k0 = 0; k0 < DD; k0 += 32) {
        v16bf a0 = load_a_frag(X, DD, m0,      k0, lane);
        v16bf a1 = load_a_frag(X, DD, m0 + 16, k0, lane);
#pragma unroll
        for (int nn = 0; nn < 4; ++nn) {
            v16bf b = load_b_frag(W, DD, n0 + nn * 16, k0, lane);
            acc0[nn] = wmma_bf16(a0, b, acc0[nn]);
            acc1[nn] = wmma_bf16(a1, b, acc1[nn]);
        }
    }

    int half = lane >> 4, nl = lane & 15;
#pragma unroll
    for (int sub = 0; sub < 2; ++sub) {
#pragma unroll
        for (int nn = 0; nn < 4; ++nn) {
#pragma unroll
            for (int r = 0; r < 8; ++r) {
                int grow = m0 + sub * 16 + r + (half << 3);
                int b    = grow >> 11;
                int s    = grow & 2047;
                int e    = n0 + nn * 16 + nl;
                int h    = e >> 6;
                int hd   = e & 63;
                float v  = sub ? acc1[nn][r] : acc0[nn][r];
                if (vmode)
                    Y[(size_t)((b * HH + h) * HD + hd) * SS + s] = (__bf16)v;
                else
                    Y[(size_t)((b * HH + h) * SS + s) * HD + hd] = (__bf16)v;
            }
        }
    }
}

// ---------------------------------------------------------------------------
// Flash-style attention, 32-query tile per wave (2 row-tiles share every
// K/rpr/V fragment).  score = Q K^T + Q rpr_m^T + causal(-1e9);
// softmax(score*SCALE) online;  O = P V with V stored transposed (B,H,HD,S).
// K/rpr fragments for the two 32-wide d-chunks reuse the same registers to
// limit peak VGPR pressure.
// ---------------------------------------------------------------------------
__global__ void attn_kernel(const __bf16* __restrict__ Q,
                            const __bf16* __restrict__ K,
                            const __bf16* __restrict__ Vt,   // (B,H,HD,S)
                            const __bf16* __restrict__ RPR,  // (B,S,HD)
                            __bf16* __restrict__ Yout) {     // (B,S,D)
    __shared__ __bf16 plds_all[8 * 32 * 32];   // 2KB per wave, 8 waves
    int wave = threadIdx.x >> 5;
    int lane = threadIdx.x & 31;
    int gw   = blockIdx.x * (blockDim.x >> 5) + wave;
    int qt = gw & 63;             // S/32 query tiles
    int h  = (gw >> 6) & 15;
    int b  = gw >> 10;
    int i0 = qt << 5;

    const __bf16* qp = Q  + (size_t)(b * HH + h) * SS * HD;
    const __bf16* kp = K  + (size_t)(b * HH + h) * SS * HD;
    const __bf16* vp = Vt + (size_t)(b * HH + h) * HD * SS;
    const __bf16* rp = RPR + (size_t)b * SS * HD;
    __bf16* lp = plds_all + wave * (32 * 32);

    v16bf aq[2][2];
#pragma unroll
    for (int mt = 0; mt < 2; ++mt) {
        aq[mt][0] = load_a_frag(qp, HD, i0 + mt * 16, 0,  lane);
        aq[mt][1] = load_a_frag(qp, HD, i0 + mt * 16, 32, lane);
    }

    int half = lane >> 4, nl = lane & 15;
    float rmax[2][8], rsum[2][8];
    v8f oacc[2][4] = {};
#pragma unroll
    for (int mt = 0; mt < 2; ++mt)
#pragma unroll
        for (int r = 0; r < 8; ++r) { rmax[mt][r] = -3.0e38f; rsum[mt][r] = 0.f; }

    for (int j0 = 0; j0 < i0 + 32; j0 += 32) {
        v8f sc[2][2];
#pragma unroll
        for (int t = 0; t < 2; ++t) {
            int jt = j0 + t * 16;
            v8f s0 = {}, s1 = {};
            {   // d-chunk 0: fragments shared by both row tiles
                v16bf fk = load_b_frag(kp, HD, jt, 0, lane);
                v16bf fr = load_b_frag(rp, HD, jt, 0, lane);
                s0 = wmma_bf16(aq[0][0], fk, s0);
                s0 = wmma_bf16(aq[0][0], fr, s0);
                s1 = wmma_bf16(aq[1][0], fk, s1);
                s1 = wmma_bf16(aq[1][0], fr, s1);
            }
            {   // d-chunk 1: same fragment registers reused
                v16bf fk = load_b_frag(kp, HD, jt, 32, lane);
                v16bf fr = load_b_frag(rp, HD, jt, 32, lane);
                s0 = wmma_bf16(aq[0][1], fk, s0);
                s0 = wmma_bf16(aq[0][1], fr, s0);
                s1 = wmma_bf16(aq[1][1], fk, s1);
                s1 = wmma_bf16(aq[1][1], fr, s1);
            }
            int j = jt + nl;
#pragma unroll
            for (int r = 0; r < 8; ++r) {
                int ib = i0 + r + (half << 3);
                if (j > ib)      s0[r] += -1.0e9f;   // additive causal mask
                if (j > ib + 16) s1[r] += -1.0e9f;
            }
            sc[0][t] = s0;
            sc[1][t] = s1;
        }

        // online softmax update per row (rows confined to 16-lane halves)
#pragma unroll
        for (int mt = 0; mt < 2; ++mt) {
#pragma unroll
            for (int r = 0; r < 8; ++r) {
                float tm = fmaxf(sc[mt][0][r], sc[mt][1][r]);
                for (int off = 1; off < 16; off <<= 1)
                    tm = fmaxf(tm, __shfl_xor(tm, off, 32));
                float nm   = fmaxf(rmax[mt][r], tm);
                float corr = __expf(ATT_SCALE * (rmax[mt][r] - nm));
                float p0   = __expf(ATT_SCALE * (sc[mt][0][r] - nm));
                float p1   = __expf(ATT_SCALE * (sc[mt][1][r] - nm));
                float ps   = p0 + p1;
                for (int off = 1; off < 16; off <<= 1)
                    ps += __shfl_xor(ps, off, 32);
                rsum[mt][r] = rsum[mt][r] * corr + ps;
                rmax[mt][r] = nm;
                sc[mt][0][r] = p0; sc[mt][1][r] = p1;
#pragma unroll
                for (int nn = 0; nn < 4; ++nn) oacc[mt][nn][r] *= corr;
            }
        }

        // repack P (C-layout) -> A-fragment layout through per-wave LDS
#pragma unroll
        for (int mt = 0; mt < 2; ++mt)
#pragma unroll
            for (int t = 0; t < 2; ++t) {
                int kk = t * 16 + nl;
#pragma unroll
                for (int r = 0; r < 8; ++r) {
                    int m = mt * 16 + r + (half << 3);
                    lp[m * 32 + kk] = (__bf16)sc[mt][t][r];
                }
            }
        asm volatile("s_wait_dscnt 0" ::: "memory");

        // O += P(32x32) * V(32x64): V-frags shared by both row tiles
#pragma unroll
        for (int mt = 0; mt < 2; ++mt) {
            v16bf pa;
            int row = mt * 16 + nl, cb = half << 4;
            const v8bf* q0 = reinterpret_cast<const v8bf*>(lp + row * 32 + cb);
            const v8bf* q1 = reinterpret_cast<const v8bf*>(lp + row * 32 + cb + 8);
#pragma unroll
            for (int i = 0; i < 8; ++i) { pa[i] = (*q0)[i]; pa[i + 8] = (*q1)[i]; }
#pragma unroll
            for (int nn = 0; nn < 4; ++nn) {
                v16bf vb = *reinterpret_cast<const v16bf*>(
                    vp + (size_t)(nn * 16 + nl) * SS + j0 + (half << 4));
                oacc[mt][nn] = wmma_bf16(pa, vb, oacc[mt][nn]);
            }
        }
    }

    // normalize and write (B,S,D) bf16 for the output projection
#pragma unroll
    for (int mt = 0; mt < 2; ++mt)
#pragma unroll
        for (int nn = 0; nn < 4; ++nn)
#pragma unroll
            for (int r = 0; r < 8; ++r) {
                int s = i0 + mt * 16 + r + (half << 3);
                int e = h * HD + nn * 16 + nl;
                Yout[(size_t)(b * SS + s) * DD + e] =
                    (__bf16)(oacc[mt][nn][r] / rsum[mt][r]);
            }
}

// ---------------------------------------------------------------------------
// Output projection: out = attn(4096x1024) * Wp^T + bias, fp32 output.
// Same 32x64 tiling as gemm_qkv.
// ---------------------------------------------------------------------------
__global__ void gemm_proj(const __bf16* __restrict__ X,
                          const __bf16* __restrict__ W,
                          const float* __restrict__ bias,
                          float* __restrict__ out) {
    int wave = threadIdx.x >> 5;
    int lane = threadIdx.x & 31;
    int gw   = blockIdx.x * (blockDim.x >> 5) + wave;
    int m0   = (gw >> 4) << 5;
    int n0   = (gw & 15) << 6;

    v8f acc0[4] = {}, acc1[4] = {};
    for (int k0 = 0; k0 < DD; k0 += 32) {
        v16bf a0 = load_a_frag(X, DD, m0,      k0, lane);
        v16bf a1 = load_a_frag(X, DD, m0 + 16, k0, lane);
#pragma unroll
        for (int nn = 0; nn < 4; ++nn) {
            v16bf b = load_b_frag(W, DD, n0 + nn * 16, k0, lane);
            acc0[nn] = wmma_bf16(a0, b, acc0[nn]);
            acc1[nn] = wmma_bf16(a1, b, acc1[nn]);
        }
    }

    int half = lane >> 4, nl = lane & 15;
#pragma unroll
    for (int nn = 0; nn < 4; ++nn) {
        int e = n0 + nn * 16 + nl;
        float bv = bias[e];
#pragma unroll
        for (int r = 0; r < 8; ++r) {
            int g0 = m0 + r + (half << 3);
            out[(size_t)g0 * DD + e]        = acc0[nn][r] + bv;
            out[(size_t)(g0 + 16) * DD + e] = acc1[nn][r] + bv;
        }
    }
}

// ---------------------------------------------------------------------------
extern "C" void kernel_launch(void* const* d_in, const int* in_sizes, int n_in,
                              void* d_out, int out_size, void* d_ws, size_t ws_size,
                              hipStream_t stream) {
    const float* x   = (const float*)d_in[0];
    const int*   pos = (const int*)  d_in[1];
    // d_in[2] = causal_mask: computed analytically in-kernel, not read
    const float* wq  = (const float*)d_in[3];
    const float* wk  = (const float*)d_in[4];
    const float* wv  = (const float*)d_in[5];
    const float* rpr = (const float*)d_in[6];
    const float* pw  = (const float*)d_in[7];
    const float* pb  = (const float*)d_in[8];
    float* out = (float*)d_out;

    char* ws = (char*)d_ws;
    size_t o = 0;
    __bf16* xbf   = (__bf16*)(ws + o); o += (size_t)MM * DD * 2;
    __bf16* wqbf  = (__bf16*)(ws + o); o += (size_t)DD * DD * 2;
    __bf16* wkbf  = (__bf16*)(ws + o); o += (size_t)DD * DD * 2;
    __bf16* wvbf  = (__bf16*)(ws + o); o += (size_t)DD * DD * 2;
    __bf16* pwbf  = (__bf16*)(ws + o); o += (size_t)DD * DD * 2;
    __bf16* qbf   = (__bf16*)(ws + o); o += (size_t)MM * DD * 2;   // (B,H,S,HD)
    __bf16* kbf   = (__bf16*)(ws + o); o += (size_t)MM * DD * 2;   // (B,H,S,HD)
    __bf16* vbf   = (__bf16*)(ws + o); o += (size_t)MM * DD * 2;   // (B,H,HD,S)
    __bf16* rprbf = (__bf16*)(ws + o); o += (size_t)BB * SS * HD * 2;
    __bf16* abf   = (__bf16*)(ws + o); o += (size_t)MM * DD * 2;
    (void)ws_size; (void)in_sizes; (void)n_in; (void)out_size;

    const int T = 256;
    cvt_f32_bf16<<<(MM * DD + T - 1) / T, T, 0, stream>>>(x,  xbf,  MM * DD);
    cvt_f32_bf16<<<(DD * DD + T - 1) / T, T, 0, stream>>>(wq, wqbf, DD * DD);
    cvt_f32_bf16<<<(DD * DD + T - 1) / T, T, 0, stream>>>(wk, wkbf, DD * DD);
    cvt_f32_bf16<<<(DD * DD + T - 1) / T, T, 0, stream>>>(wv, wvbf, DD * DD);
    cvt_f32_bf16<<<(DD * DD + T - 1) / T, T, 0, stream>>>(pw, pwbf, DD * DD);
    gather_rpr<<<(BB * SS * HD + T - 1) / T, T, 0, stream>>>(rpr, pos, rprbf,
                                                             BB * SS * HD);
    // GEMMs: 128 M-tiles * 16 N-tiles = 2048 waves -> 256 blocks of 8 waves
    gemm_qkv<<<256, T, 0, stream>>>(xbf, wqbf, qbf, 0);
    gemm_qkv<<<256, T, 0, stream>>>(xbf, wkbf, kbf, 0);
    gemm_qkv<<<256, T, 0, stream>>>(xbf, wvbf, vbf, 1);
    // attention: B*H*(S/32) = 2048 waves -> 256 blocks
    attn_kernel<<<256, T, 0, stream>>>(qbf, kbf, vbf, rprbf, abf);
    // output projection (fp32 + bias) straight into d_out
    gemm_proj<<<256, T, 0, stream>>>(abf, pwbf, pb, out);
}
